// DCF_57372173139982
// MI455X (gfx1250) — matrix operand order repaired and checked by reading, not compiled, not run on hardware
//
#include <hip/hip_runtime.h>

// ---------------------------------------------------------------------------
// MI455X (gfx1250, wave32) implementation.
// conv1 (implicit GEMM K=1728) and the final 1x1 GEMM (K=768) run on
// v_wmma_f32_16x16x32_bf16 with f32 accumulation. LDS tiles are staged with
// GLOBAL_LOAD_ASYNC_TO_LDS_B128 (ASYNCcnt) when the toolchain exposes the
// builtin, else synchronous b128 copies. Small ops stay on VALU/TRANS.
// ---------------------------------------------------------------------------

typedef unsigned short u16;
typedef __attribute__((ext_vector_type(16))) __bf16 v16bf;
typedef __attribute__((ext_vector_type(8)))  float  v8f;
typedef __attribute__((ext_vector_type(8)))  int    v8i;
typedef __attribute__((ext_vector_type(4)))  int    v4i;

#define NN   8
#define CC   128
#define CWW  64
#define CIN  192      // C + CW
#define HH   96
#define WW_  96
#define PIX  (HH*WW_)         // 9216 per image
#define K1   (9*CIN)          // 1728, conv1 reduction
#define K2   768              // C*NUM_BASES, final reduction
#define NB   6
#define TEM  6

#if defined(__has_builtin)
#if __has_builtin(__builtin_amdgcn_global_load_async_to_lds_b128)
#define ASYNC_LDS 1
#endif
#endif

__device__ __forceinline__ u16 f2bf(float f) {
  return __builtin_bit_cast(u16, (__bf16)f);       // native cvt on gfx1250
}
__device__ __forceinline__ float bf2f(u16 h) {
  unsigned int u = ((unsigned int)h) << 16;
  return __builtin_bit_cast(float, u);
}
// tanh via TRANS ops (v_exp_f32 + v_rcp_f32): co-executes with WMMA.
__device__ __forceinline__ float fast_tanh(float x) {
  float e = __expf(2.0f * x);
  return 1.0f - 2.0f * __builtin_amdgcn_rcpf(e + 1.0f);
}
__device__ __forceinline__ v4i zero4() {
  v4i v; v[0] = 0; v[1] = 0; v[2] = 0; v[3] = 0; return v;
}
__device__ __forceinline__ v16bf join16(v4i lo, v4i hi) {
  v8i t;
#pragma unroll
  for (int j = 0; j < 4; ++j) { t[j] = lo[j]; t[j + 4] = hi[j]; }
  return __builtin_bit_cast(v16bf, t);
}

// 16B global -> LDS stage (async path writes LDS directly, no VGPR bounce).
__device__ __forceinline__ void stage16(u16* dst, const u16* src) {
#ifdef ASYNC_LDS
  typedef __attribute__((address_space(1))) v4i gv4i;
  typedef __attribute__((address_space(3))) v4i lv4i;
  __builtin_amdgcn_global_load_async_to_lds_b128(
      (gv4i*)(const_cast<u16*>(src)), (lv4i*)dst, 0, 0);
#else
  *(v4i*)dst = *(const v4i*)src;
#endif
}
__device__ __forceinline__ void wait_async() {
#ifdef ASYNC_LDS
#if __has_builtin(__builtin_amdgcn_s_wait_asynccnt)
  __builtin_amdgcn_s_wait_asynccnt(0);
#else
  asm volatile("s_wait_asynccnt 0" ::: "memory");
#endif
#endif
}

// ---------------------------------------------------------------------------
// Pack kernels: f32 -> bf16 layout conversions (one-time, bandwidth trivial).
// ---------------------------------------------------------------------------

// xb: NHWC bf16, channel = [feat(0..127), weight(128..191)] contiguous per px.
__global__ __launch_bounds__(256) void k_pack_x(const float* __restrict__ feat,
                                                const float* __restrict__ wgt,
                                                u16* __restrict__ xb, int total) {
  int i = blockIdx.x * 256 + threadIdx.x;
  if (i >= total) return;
  int c = i % CIN;
  int pix = i / CIN;                // (n*PIX + hw), pixel-major
  int n = pix / PIX, hw = pix % PIX;
  float v = (c < CC) ? feat[((size_t)n * CC + c) * PIX + hw]
                     : wgt[((size_t)n * CWW + (c - CC)) * PIX + hw];
  xb[i] = f2bf(v);
}

// w1r[co][tap*192 + cin]  (tap-major K so each 32-chunk is contiguous cin)
__global__ __launch_bounds__(256) void k_pack_w1(const float* __restrict__ w1,
                                                 u16* __restrict__ w1r) {
  int i = blockIdx.x * 256 + threadIdx.x;
  if (i >= CC * K1) return;
  int co = i / K1, k = i % K1;
  int tap = k / CIN, cin = k % CIN;
  w1r[i] = f2bf(w1[((size_t)co * CIN + cin) * 9 + tap]);
}

__global__ __launch_bounds__(256) void k_pack_coef(const float* __restrict__ cf,
                                                   u16* __restrict__ cfb, int total) {
  int i = blockIdx.x * 256 + threadIdx.x;
  if (i < total) cfb[i] = f2bf(cf[i]);
}

// ---------------------------------------------------------------------------
// conv1 (3x3, 192->128) + tanh, implicit GEMM on WMMA bf16.
// Block: 8 waves = 128 out-ch, 32-pixel row segment. 54 K-chunks x 2 halves.
// ---------------------------------------------------------------------------
__global__ __launch_bounds__(256) void k_conv1(const u16* __restrict__ xb,
                                               const u16* __restrict__ w1r,
                                               const float* __restrict__ b1,
                                               u16* __restrict__ hmid) {
  __shared__ __align__(16) u16 xt[3 * 34 * CIN];   // halo tile, 39168 B
  const int w0 = blockIdx.x * 32, h = blockIdx.y, n = blockIdx.z;
  const int t = threadIdx.x;

  // Stage input halo tile (rows h-1..h+1, cols w0-1..w0+32, zero padded).
  const int NI4 = 3 * 34 * (CIN / 8);              // b128 transfers
  for (int i = t; i < NI4; i += 256) {
    int c8 = i % (CIN / 8), pos = i / (CIN / 8);
    int r = pos / 34, col = pos % 34;
    int gh = h - 1 + r, gw = w0 - 1 + col;
    u16* dst = xt + (r * 34 + col) * CIN + c8 * 8;
    if (gh >= 0 && gh < HH && gw >= 0 && gw < WW_)
      stage16(dst, xb + (((size_t)n * HH + gh) * WW_ + gw) * CIN + c8 * 8);
    else
      *(v4i*)dst = zero4();
  }
  wait_async();
  __syncthreads();

  const int lane = t & 31, wv = t >> 5;
  const int co0 = wv * 16, m = lane & 15;
  const bool hi = lane >= 16;

  v8f acc0, acc1;
#pragma unroll
  for (int v = 0; v < 8; ++v) {
    float bv = b1[co0 + v + (hi ? 8 : 0)];
    acc0[v] = bv; acc1[v] = bv;
  }

  const u16* arow = w1r + (size_t)(co0 + m) * K1 + (hi ? 8 : 0);
  for (int kc = 0; kc < 54; ++kc) {
    const int k0 = kc * 32;
    const int tap = k0 / CIN, cin0 = k0 % CIN;     // K is tap-major
    const int r = tap / 3, s = tap % 3;
    // A fragment (16x32 bf16): ISA layout — lanes<16 hold K 0-7 & 16-23.
    const u16* ap = arow + k0;
    v16bf a = join16(*(const v4i*)ap, *(const v4i*)(ap + 16));
    if (kc + 1 < 54) __builtin_prefetch((const void*)(ap + 32), 0, 1);
    // B fragments (32x16 bf16), pixels 0-15 and 16-31.
    const u16* brow = xt + (r * 34) * CIN + cin0 + (hi ? 8 : 0);
    {
      const u16* bp = brow + (size_t)(m + s) * CIN;
      v16bf b = join16(*(const v4i*)bp, *(const v4i*)(bp + 16));
      acc0 = __builtin_amdgcn_wmma_f32_16x16x32_bf16(false, a, false, b,
                                                     (short)0, acc0, false, false);
    }
    {
      const u16* bp = brow + (size_t)(m + 16 + s) * CIN;
      v16bf b = join16(*(const v4i*)bp, *(const v4i*)(bp + 16));
      acc1 = __builtin_amdgcn_wmma_f32_16x16x32_bf16(false, a, false, b,
                                                     (short)0, acc1, false, false);
    }
  }

  // tanh + packed bf16x2 stores to hmid (NHWC, channels contiguous).
#pragma unroll
  for (int half = 0; half < 2; ++half) {
    int pixw = w0 + m + half * 16;
    size_t pbase = (((size_t)n * HH + h) * WW_ + pixw) * CC;
    unsigned int* hp32 = (unsigned int*)(hmid + pbase + co0 + (hi ? 8 : 0));
#pragma unroll
    for (int v = 0; v < 8; v += 2) {
      float x0 = fast_tanh(half ? acc1[v]     : acc0[v]);
      float x1 = fast_tanh(half ? acc1[v + 1] : acc0[v + 1]);
      hp32[v / 2] = (unsigned int)f2bf(x0) | ((unsigned int)f2bf(x1) << 16);
    }
  }
}

// ---------------------------------------------------------------------------
// conv2 (1x1, 128->36) + tanh, then mix with fixed FB bases (6x6 @ 6x9).
// One thread per pixel; 0.7 GFLOP total -> VALU is the right tool.
// Output: basesb[pixel][64] bf16 (54 used, padded for aligned b128 staging).
// ---------------------------------------------------------------------------
__global__ __launch_bounds__(256) void k_conv2_bases(const u16* __restrict__ hmid,
                                                     const float* __restrict__ w2,
                                                     const float* __restrict__ b2,
                                                     const float* __restrict__ bbuf,
                                                     u16* __restrict__ basesb) {
  int q = blockIdx.x * 256 + threadIdx.x;
  if (q >= NN * PIX) return;
  const u16* hp = hmid + (size_t)q * CC;
  float bb[36];
#pragma unroll
  for (int mo = 0; mo < 36; ++mo) bb[mo] = b2[mo];
  for (int kb = 0; kb < CC / 8; ++kb) {            // b128 chunks of hmid
    v4i raw = *(const v4i*)(hp + kb * 8);
#pragma unroll
    for (int j = 0; j < 4; ++j) {
      unsigned int w = (unsigned int)raw[j];
      float x0 = bf2f((u16)(w & 0xffffu));
      float x1 = bf2f((u16)(w >> 16));
      int k = kb * 8 + j * 2;
#pragma unroll
      for (int mo = 0; mo < 36; ++mo) {
        bb[mo] += w2[mo * CC + k] * x0;
        bb[mo] += w2[mo * CC + k + 1] * x1;
      }
    }
  }
#pragma unroll
  for (int mo = 0; mo < 36; ++mo) bb[mo] = fast_tanh(bb[mo]);
  u16* op = basesb + (size_t)q * 64;
#pragma unroll
  for (int mm = 0; mm < NB; ++mm)
#pragma unroll
    for (int l = 0; l < 9; ++l) {
      float s = 0.f;
#pragma unroll
      for (int k2 = 0; k2 < TEM; ++k2) s += bb[mm * TEM + k2] * bbuf[k2 * 9 + l];
      op[mm * 9 + l] = f2bf(s);
    }
}

// ---------------------------------------------------------------------------
// Final: per-pixel bases_out (768) built in LDS (VALU, 1 GFLOP), then
// WMMA GEMM with coef (128 x 768) + bias -> f32 NCHW output.
// Block: 8 waves = 128 out-ch, 16-pixel row segment. 24 WMMA per wave.
// ---------------------------------------------------------------------------
__global__ __launch_bounds__(256) void k_final(const u16* __restrict__ xb,
                                               const u16* __restrict__ basesb,
                                               const u16* __restrict__ coefb,
                                               const float* __restrict__ bias,
                                               float* __restrict__ out) {
  __shared__ __align__(16) u16 ft[3 * 18 * CC];    // feat halo tile, 13824 B
  __shared__ __align__(16) u16 bt[16 * 64];        // bases tile,      2048 B
  __shared__ __align__(16) u16 bout[16 * K2];      // bases_out,      24576 B
  const int w0 = blockIdx.x * 16, h = blockIdx.y, n = blockIdx.z;
  const int t = threadIdx.x;

  // Stage feat patches (channels 0..127 of xb are bf16 feat).
  for (int i = t; i < 3 * 18 * (CC / 8); i += 256) {
    int c8 = i % (CC / 8), pos = i / (CC / 8);
    int r = pos / 18, col = pos % 18;
    int gh = h - 1 + r, gw = w0 - 1 + col;
    u16* dst = ft + (r * 18 + col) * CC + c8 * 8;
    if (gh >= 0 && gh < HH && gw >= 0 && gw < WW_)
      stage16(dst, xb + (((size_t)n * HH + gh) * WW_ + gw) * CIN + c8 * 8);
    else
      *(v4i*)dst = zero4();
  }
  // Stage per-pixel bases (16 px * 8 b128).
  for (int i = t; i < 16 * 8; i += 256) {
    int p = i / 8, c8 = i % 8;
    stage16(bt + p * 64 + c8 * 8,
            basesb + (size_t)(((n * HH + h) * WW_) + w0 + p) * 64 + c8 * 8);
  }
  wait_async();
  __syncthreads();

  // bases_out[p][c*6+m] = sum_l bases(p,m,l) * patch(p,c,l)   (bf16 into LDS)
  {
    int p = t >> 4, cg = t & 15;                   // pixel, channel-group of 8
    float bs[54];
#pragma unroll
    for (int j = 0; j < 54; ++j) bs[j] = bf2f(bt[p * 64 + j]);
    for (int ci = 0; ci < 8; ++ci) {
      int c = cg * 8 + ci;
      float pt[9];
#pragma unroll
      for (int r = 0; r < 3; ++r)
#pragma unroll
        for (int s = 0; s < 3; ++s)
          pt[r * 3 + s] = bf2f(ft[(r * 18 + p + s) * CC + c]);
#pragma unroll
      for (int mm = 0; mm < NB; ++mm) {
        float sum = 0.f;
#pragma unroll
        for (int l = 0; l < 9; ++l) sum += bs[mm * 9 + l] * pt[l];
        bout[p * K2 + c * NB + mm] = f2bf(sum);
      }
    }
  }
  __syncthreads();

  // GEMM: coef (128 x 768) x bases_out^T (768 x 16) on WMMA bf16.
  const int lane = t & 31, wv = t >> 5;
  const int co0 = wv * 16, m = lane & 15;
  const bool hi = lane >= 16;
  v8f acc;
#pragma unroll
  for (int v = 0; v < 8; ++v) acc[v] = bias[co0 + v + (hi ? 8 : 0)];

  const u16* arow = coefb + (size_t)(co0 + m) * K2 + (hi ? 8 : 0);
  const u16* brow = bout + (size_t)m * K2 + (hi ? 8 : 0);
  for (int kc = 0; kc < K2 / 32; ++kc) {
    const int k0 = kc * 32;
    const u16* ap = arow + k0;
    v16bf a = join16(*(const v4i*)ap, *(const v4i*)(ap + 16));
    if (kc + 1 < K2 / 32) __builtin_prefetch((const void*)(ap + 32), 0, 1);
    const u16* bp = brow + k0;
    v16bf b = join16(*(const v4i*)bp, *(const v4i*)(bp + 16));
    acc = __builtin_amdgcn_wmma_f32_16x16x32_bf16(false, a, false, b,
                                                  (short)0, acc, false, false);
  }

#pragma unroll
  for (int v = 0; v < 8; ++v) {
    int co = co0 + v + (hi ? 8 : 0);
    out[(((size_t)n * CC + co) * HH + h) * WW_ + w0 + m] = acc[v];
  }
}

// ---------------------------------------------------------------------------
// Host launcher. Workspace layout (bytes, all 16B-aligned):
//   xb     28,311,552   (NHWC bf16 concat input)
//   w1r       442,368   (conv1 weights, tap-major bf16)
//   coefb     196,608   (coef bf16)
//   hmid   18,874,368   (NHWC bf16)
//   basesb  9,437,184   (per-pixel 64-padded bf16)
//   total ~57.3 MB
// ---------------------------------------------------------------------------
extern "C" void kernel_launch(void* const* d_in, const int* in_sizes, int n_in,
                              void* d_out, int out_size, void* d_ws, size_t ws_size,
                              hipStream_t stream) {
  const float* feat = (const float*)d_in[0];
  const float* wgt  = (const float*)d_in[1];
  const float* w1   = (const float*)d_in[2];
  const float* b1   = (const float*)d_in[3];
  const float* w2   = (const float*)d_in[4];
  const float* b2   = (const float*)d_in[5];
  const float* bbuf = (const float*)d_in[6];
  const float* coef = (const float*)d_in[7];
  const float* bias = (const float*)d_in[8];
  float* out = (float*)d_out;

  char* ws = (char*)d_ws;
  u16* xb     = (u16*)(ws);
  u16* w1r    = (u16*)(ws + 28311552);
  u16* coefb  = (u16*)(ws + 28311552 + 442368);
  u16* hmid   = (u16*)(ws + 28311552 + 442368 + 196608);
  u16* basesb = (u16*)(ws + 28311552 + 442368 + 196608 + 18874368);

  const int xtot = NN * PIX * CIN;                 // 14,155,776
  k_pack_x<<<(xtot + 255) / 256, 256, 0, stream>>>(feat, wgt, xb, xtot);
  k_pack_w1<<<(CC * K1 + 255) / 256, 256, 0, stream>>>(w1, w1r);
  k_pack_coef<<<(CC * K2 + 255) / 256, 256, 0, stream>>>(coef, coefb, CC * K2);

  k_conv1<<<dim3(WW_ / 32, HH, NN), 256, 0, stream>>>(xb, w1r, b1, hmid);
  k_conv2_bases<<<(NN * PIX + 255) / 256, 256, 0, stream>>>(hmid, w2, b2, bbuf, basesb);
  k_final<<<dim3(WW_ / 16, HH, NN), 256, 0, stream>>>(xb, basesb, coefb, bias, out);
}
